// MAGNetLayer_3582002725418
// MI455X (gfx1250) — matrix operand order repaired and checked by reading, not compiled, not run on hardware
//
#include <hip/hip_runtime.h>
#include <math.h>

#define N_NODES 8192
#define IN_DIM  512
#define OUT_DIM 512
#define TASK_DIM 64
#define GATE_IN (IN_DIM + TASK_DIM)   // 576
#define NEG_INF_F (-9.0e15f)
#define LPITCH 40                     // 32 + 8 pad halves -> 80B = 5*16B

typedef __attribute__((ext_vector_type(16))) __bf16 v16bf;
typedef __attribute__((ext_vector_type(8)))  float  v8f;
typedef __attribute__((ext_vector_type(4)))  unsigned int v4u;
typedef __attribute__((ext_vector_type(8)))  int    v8i;
typedef __attribute__((ext_vector_type(4)))  int    v4i;

union Frag32 { uint4 q[2]; v16bf v; };          // 32 bytes = 16 bf16

__device__ __forceinline__ unsigned short f2bf(float f) {
  union { float f; unsigned u; } x; x.f = f;
  unsigned r = (x.u + 0x7FFFu + ((x.u >> 16) & 1u)) >> 16;  // RNE
  return (unsigned short)r;
}
__device__ __forceinline__ unsigned pack2(float a, float b) {
  return (unsigned)f2bf(a) | ((unsigned)f2bf(b) << 16);
}

// A fragment: two b128 chunks of a k-contiguous 32-half row
__device__ __forceinline__ v16bf load_afrag(const unsigned short* row, int half) {
  const uint4* rp = (const uint4*)row;          // row is 16B aligned (pitch 80B)
  Frag32 f;
  f.q[0] = rp[2 * half];                        // k = 16*half  .. +7
  f.q[1] = rp[2 * half + 1];                    // k = 16*half+8.. +15
  return f.v;
}
// B fragment: one 32B k-contiguous run of an [n][k] row
__device__ __forceinline__ v16bf load_bfrag(const unsigned short* row, int half) {
  const uint4* rp = (const uint4*)row;
  Frag32 f;
  f.q[0] = rp[2 * half];                        // k = 16*half .. +7
  f.q[1] = rp[2 * half + 1];                    // k = 16*half+8.. +15
  return f.v;
}

// ---------------- K0: fold attention vector 'a' into Wq/Wk -------------------
__global__ void k_prep(const float* __restrict__ Wk, const float* __restrict__ bk,
                       const float* __restrict__ Wq, const float* __restrict__ bq,
                       const float* __restrict__ a,
                       float* vk, float* vq, float* cvals) {
  int t = threadIdx.x;
  for (int tt = t; tt < OUT_DIM; tt += 256) {
    float sk = 0.f, sq = 0.f;
    for (int d = 0; d < OUT_DIM; ++d) {
      sk += Wk[(size_t)tt * OUT_DIM + d] * a[d];
      sq += Wq[(size_t)tt * OUT_DIM + d] * a[OUT_DIM + d];
    }
    vk[tt] = sk; vq[tt] = sq;
  }
  if (t == 0) {
    float ck = 0.f, cq = 0.f;
    for (int d = 0; d < OUT_DIM; ++d) {
      ck += bk[d] * a[d];
      cq += bq[d] * a[OUT_DIM + d];
    }
    cvals[0] = ck; cvals[1] = cq;
  }
}

// ------------- transpose+convert: src f32 [K][512] -> dst bf16 [512][K] ------
__global__ void __launch_bounds__(256) k_transpose_bf(const float* __restrict__ src,
                                                      int K,
                                                      unsigned short* __restrict__ dst) {
  __shared__ float tile[32][33];
  int tid = threadIdx.x;
  int kb = blockIdx.x * 32, nb = blockIdx.y * 32;
  #pragma unroll
  for (int r = 0; r < 4; ++r) {
    int idx = r * 256 + tid;
    int rr = idx >> 5, cc = idx & 31;           // rr: k-local, cc: n-local
    tile[rr][cc] = src[(size_t)(kb + rr) * 512 + nb + cc];
  }
  __syncthreads();
  #pragma unroll
  for (int r = 0; r < 2; ++r) {
    int idx = r * 256 + tid;
    int rr = idx >> 4, cp = idx & 15;           // rr: n-local, cp: k pair
    unsigned v = pack2(tile[2 * cp][rr], tile[2 * cp + 1][rr]);
    *(unsigned*)(&dst[(size_t)(nb + rr) * K + kb + 2 * cp]) = v;
  }
}

// ---------------- K_sk: sk[i] = h[i,:] . vk + ck (GEMV) ----------------------
__global__ void __launch_bounds__(256) k_sk(const float* __restrict__ h,
                                            const float* __restrict__ vk,
                                            const float* __restrict__ cvals,
                                            float* __restrict__ sk) {
  int wave = threadIdx.x >> 5, lane = threadIdx.x & 31;
  int i = blockIdx.x * 8 + wave;
  float s = 0.f;
  for (int c = lane; c < IN_DIM; c += 32)
    s += h[(size_t)i * IN_DIM + c] * vk[c];
  #pragma unroll
  for (int off = 16; off; off >>= 1) s += __shfl_down(s, off, 32);
  if (lane == 0) sk[i] = s + cvals[0];
}

// ---------------- K1: fused message/gate GEMM (K=576 over h||z) --------------
__global__ void __launch_bounds__(256) k_msg_gate(
    const float* __restrict__ h, const float* __restrict__ z,
    const unsigned short* __restrict__ Wmt,   // bf16 [512][512]
    const float* __restrict__ bmv,
    const unsigned short* __restrict__ Wgt,   // bf16 [512][576]
    const float* __restrict__ bgv,
    const float* __restrict__ vq, const float* __restrict__ cvals,
    unsigned short* __restrict__ Gt,          // bf16 [512][8192]
    float* __restrict__ sq, float* __restrict__ l1part) {
  __shared__ __align__(16) unsigned short sA[32][LPITCH];
  __shared__ __align__(16) unsigned short sWt[512][LPITCH];
  __shared__ float s_sqa[32];
  __shared__ float red[256];

  int tid = threadIdx.x, lane = tid & 31, wave = tid >> 5;
  int waveM = wave >> 2, waveN = wave & 3;
  int half = lane >> 4;
  int ib = blockIdx.x * 32;

  if (tid < 32) s_sqa[tid] = 0.f;

  v8f accM[8], accG[8];
  #pragma unroll
  for (int t = 0; t < 8; ++t) {
    #pragma unroll
    for (int q = 0; q < 8; ++q) { accM[t][q] = 0.f; accG[t][q] = 0.f; }
  }

  for (int k0 = 0; k0 < GATE_IN; k0 += 32) {
    bool hasM = (k0 < IN_DIM);
    __syncthreads();
    // A tile: rows ib..ib+31, cols k0..k0+31 (h for k<512, z beyond)
    #pragma unroll
    for (int r = 0; r < 2; ++r) {
      int idx = r * 256 + tid;
      int rr = idx >> 4, cp = idx & 15;
      int gk = k0 + 2 * cp;
      float2 v = (gk < IN_DIM)
          ? *(const float2*)(&h[(size_t)(ib + rr) * IN_DIM + gk])
          : *(const float2*)(&z[(size_t)(ib + rr) * TASK_DIM + (gk - IN_DIM)]);
      *(unsigned*)(&sA[rr][2 * cp]) = pack2(v.x, v.y);
    }
    // Wg tile: [n][k] uint4 copy
    #pragma unroll
    for (int rep = 0; rep < 8; ++rep) {
      int idx = rep * 256 + tid;
      int n = idx >> 2, c4 = idx & 3;
      ((uint4*)(&sWt[n][0]))[c4] =
          ((const uint4*)Wgt)[(size_t)n * (GATE_IN / 8) + (k0 >> 3) + c4];
    }
    __syncthreads();

    v16bf afrag = load_afrag(&sA[waveM * 16 + (lane & 15)][0], half);

    #pragma unroll
    for (int tb = 0; tb < 8; ++tb) {
      int n = waveN * 128 + tb * 16 + (lane & 15);
      v16bf bfr = load_bfrag(&sWt[n][0], half);
      accG[tb] = __builtin_amdgcn_wmma_f32_16x16x32_bf16(
          false, afrag, false, bfr, (short)0, accG[tb], false, false);
    }

    if (hasM) {
      __syncthreads();
      #pragma unroll
      for (int rep = 0; rep < 8; ++rep) {
        int idx = rep * 256 + tid;
        int n = idx >> 2, c4 = idx & 3;
        ((uint4*)(&sWt[n][0]))[c4] =
            ((const uint4*)Wmt)[(size_t)n * (IN_DIM / 8) + (k0 >> 3) + c4];
      }
      __syncthreads();
      #pragma unroll
      for (int tb = 0; tb < 8; ++tb) {
        int n = waveN * 128 + tb * 16 + (lane & 15);
        v16bf bfr = load_bfrag(&sWt[n][0], half);
        accM[tb] = __builtin_amdgcn_wmma_f32_16x16x32_bf16(
            false, afrag, false, bfr, (short)0, accM[tb], false, false);
      }
    }
  }

  // fused epilogue: one 16B transposed store per tile per lane
  float l1 = 0.f;
  int gi_base = ib + waveM * 16 + 8 * half;     // 8 consecutive rows per lane
  #pragma unroll
  for (int tb = 0; tb < 8; ++tb) {
    int n = waveN * 128 + tb * 16 + (lane & 15);
    float bmn = bmv[n], bgn = bgv[n], vqn = vq[n];
    union { unsigned short us[8]; uint4 q; } pk;
    #pragma unroll
    for (int r = 0; r < 8; ++r) {
      int row_local = waveM * 16 + r + 8 * half;
      float g = 1.f / (1.f + __expf(-(accG[tb][r] + bgn)));
      float mpre = accM[tb][r] + bmn;
      float msg = mpre > 0.f ? mpre : 0.f;
      float gm = g * msg;
      pk.us[r] = f2bf(gm);
      l1 += g;
      atomicAdd(&s_sqa[row_local], gm * vqn);   // ds_add_f32
    }
    ((uint4*)Gt)[(size_t)n * (N_NODES / 8) + (gi_base >> 3)] = pk.q;
  }
  red[tid] = l1;
  __syncthreads();
  for (int off = 128; off; off >>= 1) {
    if (tid < off) red[tid] += red[tid + off];
    __syncthreads();
  }
  if (tid == 0) l1part[blockIdx.x] = red[0];
  if (tid < 32) sq[ib + tid] = s_sqa[tid] + cvals[1];
}

// ---------------- K2a: online masked softmax stats per row -------------------
__global__ void __launch_bounds__(256) k_stats(const int* __restrict__ adj,
                                               const float* __restrict__ sq,
                                               const float* __restrict__ sk,
                                               float* __restrict__ mrow,
                                               float* __restrict__ sinv) {
  __shared__ float2 red[256];
  int i = blockIdx.x, tid = threadIdx.x;
  const int* arow = adj + (size_t)i * N_NODES;
  float sqi = sq[i];
  // masked entries contribute exp(NEG_INF - m) == 0 in f32: skip them entirely.
  float mloc = NEG_INF_F, sloc = 0.f;
  for (int j = tid; j < N_NODES; j += 256) {
    if (j + 2048 < N_NODES) __builtin_prefetch(arow + j + 2048, 0, 0);
    if (arow[j] > 0) {
      float e = sqi + sk[j];
      e = e > 0.f ? e : 0.01f * e;               // leaky_relu(0.01)
      float d = e - mloc;
      if (d > 0.f) { sloc = sloc * __expf(-d) + 1.f; mloc = e; }
      else sloc += __expf(d);
    }
  }
  red[tid] = make_float2(mloc, sloc);
  __syncthreads();
  for (int off = 128; off; off >>= 1) {
    if (tid < off) {
      float2 A = red[tid], B = red[tid + off];
      float M = fmaxf(A.x, B.x);
      red[tid] = make_float2(M, A.y * __expf(A.x - M) + B.y * __expf(B.x - M));
    }
    __syncthreads();
  }
  if (tid == 0) { mrow[i] = red[0].x; sinv[i] = 1.f / red[0].y; }
}

// ---------------- K2b: agg = P^T @ gated, fused h_new = (agg+h)/2 ------------
// B tile staged by the Tensor Data Mover: one TENSOR_LOAD_TO_LDS per K-chunk
// (tile 32 k-elems x 512 n-rows, TDM pad recreates the 80B LDS pitch), waited
// with s_wait_tensorcnt. A tile (P^T) synthesized from adj concurrently.
__global__ void __launch_bounds__(256) k_agg(const int* __restrict__ adj,
                                             const float* __restrict__ sq,
                                             const float* __restrict__ sk,
                                             const float* __restrict__ mrow,
                                             const float* __restrict__ sinv,
                                             const unsigned short* __restrict__ Gt,
                                             const float* __restrict__ h,
                                             float* __restrict__ out) {
  __shared__ __align__(16) unsigned short sA[32][LPITCH];     // [m=j][k=i]
  __shared__ __align__(16) unsigned short sBt[512][LPITCH];   // [n=d][k=i]

  int tid = threadIdx.x, lane = tid & 31, wave = tid >> 5;
  int waveM = wave >> 2, waveN = wave & 3;
  int half = lane >> 4;
  int jb = blockIdx.x * 32;
  int jj = tid & 31, ib4 = (tid >> 5) * 4;      // A-gen assignment
  float skj = sk[jb + jj];

  // --- loop-invariant TDM descriptor group 1 (ISA 8.4) ---
  // data_size=2B; pad_enable; pad_interval=16 DW (=tile row 64B); pad=4 DW (16B)
  // tensor_dim0=8192, tensor_dim1=512, tile_dim0=32, tile_dim1=512, stride0=8192
  const v8i g1 = { (1 << 16) | (1 << 20) | (3 << 22) | (3 << 25),
                   (int)(8192u << 16),          // tensor_dim0[15:0]
                   (int)(512u << 16),           // tensor_dim0[31:16]=0 | tensor_dim1[15:0]
                   (int)(32u << 16),            // tensor_dim1[31:16]=0 | tile_dim0
                   512,                         // tile_dim1 | tile_dim2=0
                   8192,                        // tensor_dim0_stride[31:0]
                   0, 0 };
  const v4i gz = { 0, 0, 0, 0 };
  unsigned lds_base = (unsigned)(size_t)(void*)&sBt[0][0];    // LDS byte offset

  v8f acc[8];
  #pragma unroll
  for (int t = 0; t < 8; ++t) {
    #pragma unroll
    for (int q = 0; q < 8; ++q) acc[t][q] = 0.f;
  }

  for (int k0 = 0; k0 < N_NODES; k0 += 32) {
    __syncthreads();
    // TDM: async-load B tile (bf16 Gt rows, 32 k x 512 n) into padded LDS
    if (wave == 0) {
      unsigned long long ga =
          (unsigned long long)(size_t)Gt + (unsigned long long)k0 * 2ull;
      v4u g0;
      g0.x = 1u;                                  // count=1 (valid user D#)
      g0.y = lds_base;
      g0.z = (unsigned)(ga & 0xFFFFFFFFull);
      g0.w = (unsigned)((ga >> 32) & 0x01FFFFFFull) | (2u << 30);  // type=2
#if __clang_major__ >= 23
      v8i g4 = { 0, 0, 0, 0, 0, 0, 0, 0 };
      __builtin_amdgcn_tensor_load_to_lds(g0, g1, gz, gz, g4, 0);
#else
      __builtin_amdgcn_tensor_load_to_lds(g0, g1, gz, gz, 0);
#endif
    }
    // A tile: P[i,j] for i in [k0,k0+32), j in [jb,jb+32); 4 k-contiguous
    // values per thread packed into one 8B LDS store (overlaps the TDM).
    {
      float p[4];
      #pragma unroll
      for (int u = 0; u < 4; ++u) {
        int i = k0 + ib4 + u;
        int av = adj[(size_t)i * N_NODES + jb + jj];   // coalesced over jj
        float e = sq[i] + skj;
        e = e > 0.f ? e : 0.01f * e;
        p[u] = (av > 0) ? __expf(e - mrow[i]) * sinv[i] : 0.f;
      }
      *(uint2*)(&sA[jj][ib4]) = make_uint2(pack2(p[0], p[1]), pack2(p[2], p[3]));
    }
    if (wave == 0) __builtin_amdgcn_s_wait_tensorcnt(0);
    __syncthreads();

    v16bf afrag = load_afrag(&sA[waveM * 16 + (lane & 15)][0], half);
    #pragma unroll
    for (int tb = 0; tb < 8; ++tb) {
      int n = waveN * 128 + tb * 16 + (lane & 15);
      v16bf bfr = load_bfrag(&sBt[n][0], half);
      acc[tb] = __builtin_amdgcn_wmma_f32_16x16x32_bf16(
          false, afrag, false, bfr, (short)0, acc[tb], false, false);
    }
  }

  // fused residual epilogue: h_new = (agg + h) / 2
  #pragma unroll
  for (int tb = 0; tb < 8; ++tb) {
    int n = waveN * 128 + tb * 16 + (lane & 15);
    #pragma unroll
    for (int r = 0; r < 8; ++r) {
      int ml = r + 8 * half;
      size_t o = (size_t)(jb + waveM * 16 + ml) * OUT_DIM + n;
      out[o] = (acc[tb][r] + h[o]) * 0.5f;
    }
  }
}

// ---------------- K3: deterministic L1-loss finalize -------------------------
__global__ void k_finalize(const float* __restrict__ l1part, float* __restrict__ out) {
  if (threadIdx.x == 0 && blockIdx.x == 0) {
    float s = 0.f;
    for (int i = 0; i < N_NODES / 32; ++i) s += l1part[i];
    out[(size_t)N_NODES * OUT_DIM] = s / ((float)N_NODES * (float)OUT_DIM);
  }
}

extern "C" void kernel_launch(void* const* d_in, const int* in_sizes, int n_in,
                              void* d_out, int out_size, void* d_ws, size_t ws_size,
                              hipStream_t stream) {
  (void)in_sizes; (void)n_in; (void)out_size; (void)ws_size;
  const float* h   = (const float*)d_in[0];
  const int*   adj = (const int*)d_in[1];
  const float* z   = (const float*)d_in[2];
  const float* Wm  = (const float*)d_in[3];
  const float* bm  = (const float*)d_in[4];
  const float* Wg  = (const float*)d_in[5];
  const float* bg  = (const float*)d_in[6];
  const float* Wk  = (const float*)d_in[7];
  const float* bk  = (const float*)d_in[8];
  const float* Wq  = (const float*)d_in[9];
  const float* bq  = (const float*)d_in[10];
  const float* a   = (const float*)d_in[11];
  float* out = (float*)d_out;

  // workspace layout (~9.6 MB), all segments 16B aligned
  char* ws = (char*)d_ws;
  unsigned short* Gt  = (unsigned short*)ws;                 // bf16 [512][8192]
  size_t off = (size_t)OUT_DIM * N_NODES * 2;
  unsigned short* Wmt = (unsigned short*)(ws + off); off += (size_t)OUT_DIM * IN_DIM * 2;
  unsigned short* Wgt = (unsigned short*)(ws + off); off += (size_t)OUT_DIM * GATE_IN * 2;
  float* vk   = (float*)(ws + off); off += OUT_DIM * 4;
  float* vq   = (float*)(ws + off); off += OUT_DIM * 4;
  float* cv   = (float*)(ws + off); off += 4 * 4;
  float* skv  = (float*)(ws + off); off += (size_t)N_NODES * 4;
  float* sqv  = (float*)(ws + off); off += (size_t)N_NODES * 4;
  float* mro  = (float*)(ws + off); off += (size_t)N_NODES * 4;
  float* sin_ = (float*)(ws + off); off += (size_t)N_NODES * 4;
  float* l1p  = (float*)(ws + off); off += (N_NODES / 32) * 4;

  k_prep<<<dim3(1), dim3(256), 0, stream>>>(Wk, bk, Wq, bq, a, vk, vq, cv);
  k_transpose_bf<<<dim3(IN_DIM / 32, OUT_DIM / 32), dim3(256), 0, stream>>>(Wm, IN_DIM, Wmt);
  k_transpose_bf<<<dim3(GATE_IN / 32, OUT_DIM / 32), dim3(256), 0, stream>>>(Wg, GATE_IN, Wgt);
  k_sk<<<dim3(N_NODES / 8), dim3(256), 0, stream>>>(h, vk, cv, skv);
  k_msg_gate<<<dim3(N_NODES / 32), dim3(256), 0, stream>>>(
      h, z, Wmt, bm, Wgt, bg, vq, cv, Gt, sqv, l1p);
  k_stats<<<dim3(N_NODES), dim3(256), 0, stream>>>(adj, sqv, skv, mro, sin_);
  k_agg<<<dim3(N_NODES / 32), dim3(256), 0, stream>>>(
      adj, sqv, skv, mro, sin_, Gt, h, out);
  k_finalize<<<dim3(1), dim3(1), 0, stream>>>(l1p, out);
}
